// BasicMPNN_51170240364727
// MI455X (gfx1250) — matrix operand order repaired and checked by reading, not compiled, not run on hardware
//
#include <hip/hip_runtime.h>

// Problem constants (match reference)
#define NN 100000
#define NE 1600000
#define FIN 16
#define HID 64
#define NL 3
#define NG 64

typedef __attribute__((ext_vector_type(16))) __bf16 v16bf;
typedef __attribute__((ext_vector_type(8)))  float  v8f;

union ABfrag { v16bf v; unsigned int u[8]; uint4 q[2]; };

// round-to-nearest-even f32 -> bf16 (bit manipulation; no scalar __bf16 math)
static __device__ __forceinline__ unsigned short f2bf(float a) {
    unsigned int ua = __builtin_bit_cast(unsigned int, a);
    return (unsigned short)((ua + 0x7FFFu + ((ua >> 16) & 1u)) >> 16);
}
static __device__ __forceinline__ unsigned int pack2bf(float a, float b) {
    return (unsigned int)f2bf(a) | ((unsigned int)f2bf(b) << 16);
}

__global__ void k_zero(float* __restrict__ p, long long n) {
    long long i = (long long)blockIdx.x * blockDim.x + threadIdx.x;
    if (i < n) p[i] = 0.0f;
}

// h = x @ W_embed + b_embed  (tiny: 0.2 GFLOP) ; also emit bf16 copy
__global__ void k_embed(const float* __restrict__ x, const float* __restrict__ W,
                        const float* __restrict__ b, float* __restrict__ h,
                        unsigned short* __restrict__ hbf) {
    int idx = blockIdx.x * 256 + threadIdx.x;
    if (idx >= NN * HID) return;
    int i = idx >> 6, j = idx & 63;
    const float* xr = x + i * FIN;
    float s = b[j];
#pragma unroll
    for (int k = 0; k < FIN; k++) s += xr[k] * W[k * HID + j];
    h[idx] = s;
    hbf[idx] = f2bf(s);
}

// Pre-swizzle weights into the 16x16x32 bf16 B-fragment lane layout so the GEMM
// kernels load each lane's 16 bf16 operand contiguously (2x b128 per fragment).
// Fragment layout: [layer][kchunk][ntile][lane(32)][elem(16)]
// lane L holds column n = ntile*16 + (L&15); elem j holds K = kc*32 + 8*(L>>4) + (j&7) + 16*(j>>3).
__global__ void k_swizzleW(const float* __restrict__ W, unsigned short* __restrict__ out,
                           int kchunks, int Ktot) {
    int idx = blockIdx.x * 256 + threadIdx.x;
    int total = NL * kchunks * 4 * 512;
    if (idx >= total) return;
    int j = idx & 15, lane = (idx >> 4) & 31, nt = (idx >> 9) & 3, kcl = idx >> 11;
    int kc = kcl % kchunks, l = kcl / kchunks;
    int n = nt * 16 + (lane & 15);
    int k = kc * 32 + (lane >> 4) * 8 + (j & 7) + (j >> 3) * 16;
    out[idx] = f2bf(W[((size_t)l * Ktot + k) * HID + n]);
}

#define WMMA_BF16(A, B, C) \
    __builtin_amdgcn_wmma_f32_16x16x32_bf16(false, (A), false, (B), (short)0, (C), false, false)

// Edge-message GEMM: per block, build A = [h[send] | h[rec] | e] for 32 edges
// (two 16x192 bf16 M-tiles) in LDS. Each of 4 waves owns one 16-col N-tile and
// runs TWO independent accumulator chains (one per M-tile) over 6 K-chunks of
// v_wmma_f32_16x16x32_bf16 — B fragments live in registers and are reused by
// both chains; independent chains keep the XDL pipe busy. Results scatter-add
// into agg[rec] with global f32 atomics.
__global__ __launch_bounds__(128) void k_msg(
    const unsigned short* __restrict__ hbf,
    const float* __restrict__ edge_attr,
    const int* __restrict__ eidx,            // [2*NE] (send rows, then rec rows)
    const float* __restrict__ W_edge, const float* __restrict__ b_edge,
    const unsigned short* __restrict__ Wsw,  // swizzled msg_W for this layer
    const float* __restrict__ bias,          // msg_b for this layer
    float* __restrict__ agg) {
    __shared__ unsigned int Au[32 * 100];    // 32 rows x 96 dwords (192 bf16), pad->100
    __shared__ int sidx[64];                 // [0..31]=send, [32..63]=rec
    const int tid = threadIdx.x;
    const int base = blockIdx.x * 32;

    if (tid < 32)       sidx[tid] = eidx[base + tid];
    else if (tid < 64)  sidx[tid] = eidx[NE + base + (tid - 32)];
    __syncthreads();

    {   // cooperative LDS fill: 8 threads per edge row, 12 dwords each, 2 rows/thread
        int part = tid & 7;
        for (int e = tid >> 3; e < 32; e += 16) {
            float ea = edge_attr[base + e];
            const unsigned int* hs = (const unsigned int*)(hbf + (size_t)sidx[e] * HID);
            const unsigned int* hr = (const unsigned int*)(hbf + (size_t)sidx[32 + e] * HID);
            for (int d = part; d < 96; d += 8) {
                unsigned int val;
                if (d < 32)       val = hs[d];                   // cols   0..63 : h[send]
                else if (d < 64)  val = hr[d - 32];              // cols  64..127: h[rec]
                else {                                           // cols 128..191: e (on the fly)
                    int c0 = (d - 64) * 2;
                    val = pack2bf(ea * W_edge[c0] + b_edge[c0],
                                  ea * W_edge[c0 + 1] + b_edge[c0 + 1]);
                }
                Au[e * 100 + d] = val;
            }
        }
    }
    __syncthreads();

    const int w = tid >> 5, lane = tid & 31;
    const int m = lane & 15, half = lane >> 4;
    const v16bf* B = (const v16bf*)Wsw;

    ABfrag bfr[6];
#pragma unroll
    for (int kc = 0; kc < 6; kc++) bfr[kc].v = B[(kc * 4 + w) * 32 + lane];

    v8f c0, c1;
#pragma unroll
    for (int r = 0; r < 8; r++) { c0[r] = 0.0f; c1[r] = 0.0f; }

#pragma unroll
    for (int kc = 0; kc < 6; kc++) {
        ABfrag a0, a1;
        const uint4* p0 = (const uint4*)&Au[m * 100 + kc * 16 + half * 4];
        const uint4* p1 = (const uint4*)&Au[(m + 16) * 100 + kc * 16 + half * 4];
        a0.q[0] = p0[0]; a0.q[1] = p0[2];
        a1.q[0] = p1[0]; a1.q[1] = p1[2];
        c0 = WMMA_BF16(a0.v, bfr[kc].v, c0);   // two independent accumulator
        c1 = WMMA_BF16(a1.v, bfr[kc].v, c1);   // chains per wave
    }
    const int n = w * 16 + m;                  // output column
    const float bv = bias[n];
#pragma unroll
    for (int r = 0; r < 8; r++) {
        int M = r + 8 * half;                  // rows of tile0 / tile1
        atomicAdd(agg + (size_t)sidx[32 + M] * HID + n,      c0[r] + bv);
        atomicAdd(agg + (size_t)sidx[32 + M + 16] * HID + n, c1[r] + bv);
    }
}

// Node update: out = [h | agg] @ upd_W + b ; h += relu(out) ; refresh bf16 copy.
// Same dual M-tile / dual accumulator structure (32 nodes per block).
__global__ __launch_bounds__(128) void k_upd(
    float* __restrict__ h, unsigned short* __restrict__ hbf,
    const float* __restrict__ agg,
    const unsigned short* __restrict__ Wsw, const float* __restrict__ bias) {
    __shared__ unsigned int Au[32 * 68];      // 32 rows x 64 dwords (128 bf16), pad->68
    const int tid = threadIdx.x;
    const int base = blockIdx.x * 32;
    {
        int part = tid & 7;
        for (int nr = tid >> 3; nr < 32; nr += 16) {
            const float* hrow = h   + (size_t)(base + nr) * HID;
            const float* arow = agg + (size_t)(base + nr) * HID;
            for (int d = part; d < 64; d += 8) {
                unsigned int val;
                if (d < 32) { int c0 = d * 2;        val = pack2bf(hrow[c0], hrow[c0 + 1]); }
                else        { int c0 = (d - 32) * 2; val = pack2bf(arow[c0], arow[c0 + 1]); }
                Au[nr * 68 + d] = val;
            }
        }
    }
    __syncthreads();

    const int w = tid >> 5, lane = tid & 31;
    const int m = lane & 15, half = lane >> 4;
    const v16bf* B = (const v16bf*)Wsw;

    ABfrag bfr[4];
#pragma unroll
    for (int kc = 0; kc < 4; kc++) bfr[kc].v = B[(kc * 4 + w) * 32 + lane];

    v8f c0, c1;
#pragma unroll
    for (int r = 0; r < 8; r++) { c0[r] = 0.0f; c1[r] = 0.0f; }

#pragma unroll
    for (int kc = 0; kc < 4; kc++) {
        ABfrag a0, a1;
        const uint4* p0 = (const uint4*)&Au[m * 68 + kc * 16 + half * 4];
        const uint4* p1 = (const uint4*)&Au[(m + 16) * 68 + kc * 16 + half * 4];
        a0.q[0] = p0[0]; a0.q[1] = p0[2];
        a1.q[0] = p1[0]; a1.q[1] = p1[2];
        c0 = WMMA_BF16(a0.v, bfr[kc].v, c0);
        c1 = WMMA_BF16(a1.v, bfr[kc].v, c1);
    }
    const int n = w * 16 + m;
    const float bv = bias[n];
#pragma unroll
    for (int r = 0; r < 8; r++) {
        int M = r + 8 * half;
        size_t off0 = (size_t)(base + M) * HID + n;
        size_t off1 = (size_t)(base + M + 16) * HID + n;
        float nh0 = h[off0] + fmaxf(c0[r] + bv, 0.0f);  // residual + relu, f32 master
        float nh1 = h[off1] + fmaxf(c1[r] + bv, 0.0f);
        h[off0] = nh0;  hbf[off0] = f2bf(nh0);
        h[off1] = nh1;  hbf[off1] = f2bf(nh1);
    }
}

__global__ void k_pool(const float* __restrict__ h, const int* __restrict__ batch,
                       float* __restrict__ pool) {
    int idx = blockIdx.x * 256 + threadIdx.x;
    if (idx >= NN * HID) return;
    int i = idx >> 6, j = idx & 63;
    atomicAdd(pool + (size_t)batch[i] * HID + j, h[idx]);
}

__global__ void k_pred(const float* __restrict__ pool, const float* __restrict__ Wp,
                       const float* __restrict__ bp, float* __restrict__ out) {
    int g = threadIdx.x;
    if (g >= NG) return;
    float s = bp[0];
#pragma unroll
    for (int j = 0; j < HID; j++) s += pool[g * HID + j] * Wp[j];
    out[g] = s;
}

extern "C" void kernel_launch(void* const* d_in, const int* in_sizes, int n_in,
                              void* d_out, int out_size, void* d_ws, size_t ws_size,
                              hipStream_t stream) {
    (void)in_sizes; (void)n_in; (void)out_size; (void)ws_size;
    const float* x         = (const float*)d_in[0];
    const float* edge_attr = (const float*)d_in[1];
    const int*   edge_idx  = (const int*)  d_in[2];
    const int*   batch     = (const int*)  d_in[3];
    const float* W_embed   = (const float*)d_in[4];
    const float* b_embed   = (const float*)d_in[5];
    const float* W_edge    = (const float*)d_in[6];
    const float* b_edge    = (const float*)d_in[7];
    const float* msg_W     = (const float*)d_in[8];
    const float* msg_b     = (const float*)d_in[9];
    const float* upd_W     = (const float*)d_in[10];
    const float* upd_b     = (const float*)d_in[11];
    const float* W_pred    = (const float*)d_in[12];
    const float* b_pred    = (const float*)d_in[13];
    float* out = (float*)d_out;

    // workspace carve-up (~64.1 MB total, all 256B-aligned slices)
    char* ws = (char*)d_ws;
    float* h              = (float*)ws;          ws += (size_t)NN * HID * 4;
    float* agg            = (float*)ws;          ws += (size_t)NN * HID * 4;
    unsigned short* hbf   = (unsigned short*)ws; ws += (size_t)NN * HID * 2;
    unsigned short* msgWs = (unsigned short*)ws; ws += (size_t)NL * 6 * 4 * 512 * 2;
    unsigned short* updWs = (unsigned short*)ws; ws += (size_t)NL * 4 * 4 * 512 * 2;
    float* pool           = (float*)ws;          ws += (size_t)NG * HID * 4;

    k_embed<<<(NN * HID + 255) / 256, 256, 0, stream>>>(x, W_embed, b_embed, h, hbf);
    k_swizzleW<<<(NL * 6 * 4 * 512 + 255) / 256, 256, 0, stream>>>(msg_W, msgWs, 6, 192);
    k_swizzleW<<<(NL * 4 * 4 * 512 + 255) / 256, 256, 0, stream>>>(upd_W, updWs, 4, 128);

    for (int l = 0; l < NL; l++) {
        k_zero<<<(NN * HID + 255) / 256, 256, 0, stream>>>(agg, (long long)NN * HID);
        k_msg<<<NE / 32, 128, 0, stream>>>(hbf, edge_attr, edge_idx, W_edge, b_edge,
                                           msgWs + (size_t)l * 6 * 4 * 512,
                                           msg_b + (size_t)l * HID, agg);
        k_upd<<<NN / 32, 128, 0, stream>>>(h, hbf, agg,
                                           updWs + (size_t)l * 4 * 4 * 512,
                                           upd_b + (size_t)l * HID);
    }
    k_zero<<<(NG * HID + 255) / 256, 256, 0, stream>>>(pool, (long long)NG * HID);
    k_pool<<<(NN * HID + 255) / 256, 256, 0, stream>>>(h, batch, pool);
    k_pred<<<1, 64, 0, stream>>>(pool, W_pred, b_pred, out);
}
